// EnhancedMultiheadAttention_27212912787824
// MI455X (gfx1250) — compile-verified, matrix operand order checked by reading, not run on hardware
//
#include <hip/hip_runtime.h>

typedef __attribute__((ext_vector_type(16))) __bf16 v16bf;
typedef __attribute__((ext_vector_type(8)))  __bf16 v8bf;
typedef __attribute__((ext_vector_type(8)))  float  v8f;

#define WMMA_BF16(a,b,c) \
  __builtin_amdgcn_wmma_f32_16x16x32_bf16(false,(a),false,(b),(short)0,(c),false,false)

// 16-element bf16 A/B fragment for v_wmma_*_16x16x32 from a K-contiguous row.
// Per ISA 7.12.2: lane half g holds K = kbase + g*8 + [0..7] and
// K = kbase + 16 + g*8 + [0..7]  ->  two aligned ds_load_b128.
__device__ __forceinline__ v16bf ld_frag(const __bf16* row, int g, int kbase) {
  v8bf lo = *reinterpret_cast<const v8bf*>(row + kbase + g * 8);
  v8bf hi = *reinterpret_cast<const v8bf*>(row + kbase + 16 + g * 8);
  return __builtin_shufflevector(lo, hi, 0,1,2,3,4,5,6,7,8,9,10,11,12,13,14,15);
}

// CDNA5 async copy global -> LDS (16B per lane), tracked by ASYNCcnt.
__device__ __forceinline__ void async_b128(unsigned lds_off,
                                           unsigned long long sbase,
                                           unsigned goff) {
  asm volatile("global_load_async_to_lds_b128 %0, %1, %2"
               :: "v"(lds_off), "v"(goff), "s"(sbase) : "memory");
}
__device__ __forceinline__ void wait_async0() {
  asm volatile("s_wait_asynccnt 0x0" ::: "memory");
}

// ---------------------------------------------------------------------------
// Pre-pass A: x (f32) -> xbf (bf16), same [4096x1024] layout.
// ---------------------------------------------------------------------------
__global__ __launch_bounds__(256) void convert_x(const float* __restrict__ x,
                                                 __bf16* __restrict__ xbf) {
  int base = blockIdx.x * 1024 + threadIdx.x;
  #pragma unroll
  for (int k = 0; k < 4; ++k)
    xbf[base + k * 256] = (__bf16)x[base + k * 256];
}

// ---------------------------------------------------------------------------
// Pre-pass B: W (f32 [k][n]) -> Wt (bf16 [n][k]) via 64x64 LDS tile.
// z selects which of the four weight matrices.
// ---------------------------------------------------------------------------
__global__ __launch_bounds__(256) void transpose_w(
    const float* __restrict__ Wq, const float* __restrict__ Wk,
    const float* __restrict__ Wv, const float* __restrict__ Wo,
    __bf16* __restrict__ Wqt, __bf16* __restrict__ Wkt,
    __bf16* __restrict__ Wvt, __bf16* __restrict__ Wot) {
  __shared__ float t[64][65];
  const int z = blockIdx.z;
  const float* src = (z == 0) ? Wq : (z == 1) ? Wk : (z == 2) ? Wv : Wo;
  __bf16* dst      = (z == 0) ? Wqt : (z == 1) ? Wkt : (z == 2) ? Wvt : Wot;
  const int k0 = blockIdx.x * 64, n0 = blockIdx.y * 64, tid = threadIdx.x;
  for (int i = tid; i < 4096; i += 256)
    t[i >> 6][i & 63] = src[(size_t)(k0 + (i >> 6)) * 1024 + n0 + (i & 63)];
  __syncthreads();
  for (int i = tid; i < 4096; i += 256)
    dst[(size_t)(n0 + (i >> 6)) * 1024 + k0 + (i & 63)] = (__bf16)t[i & 63][i >> 6];
}

// ---------------------------------------------------------------------------
// Kernel 1: QKV projection + bias + RoPE.  bf16 tiles arrive via async->LDS,
// double buffered (1 barrier / k-step).  Q scaled by Dh^-0.5.
// Q,K stored [B,H,L,Dh]; V stored transposed [B,H,Dh,L] for the PV stage.
// ---------------------------------------------------------------------------
__global__ __launch_bounds__(256) void qkv_rope_gemm(
    const __bf16* __restrict__ xbf,
    const __bf16* __restrict__ Wqt, const float* __restrict__ bq,
    const __bf16* __restrict__ Wkt, const float* __restrict__ bk,
    const __bf16* __restrict__ Wvt, const float* __restrict__ bv,
    __bf16* __restrict__ Qb, __bf16* __restrict__ Kb, __bf16* __restrict__ Vb)
{
  __shared__ __bf16 Xs[2][128 * 32];   // [m][k] bf16
  __shared__ __bf16 Ws[2][64 * 32];    // [n][k] bf16
  __shared__ float  Cs[128 * 64];

  const int tid = threadIdx.x;
  const int wv = tid >> 5, lane = tid & 31, g = lane >> 4, ln = lane & 15;
  const int wm = wv & 3, wn = wv >> 2;
  const int m0 = blockIdx.x * 128;
  const int h  = blockIdx.y;
  const int n0 = h * 64;
  const int z  = blockIdx.z;

  const __bf16* Wt = (z == 0) ? Wqt : (z == 1) ? Wkt : Wvt;
  const float*  bi = (z == 0) ? bq  : (z == 1) ? bk  : bv;
  const unsigned long long xA = (unsigned long long)(uintptr_t)xbf;
  const unsigned long long wA = (unsigned long long)(uintptr_t)Wt;

  auto issue = [&](int k0, int buf) {
    unsigned xl = (unsigned)(uintptr_t)&Xs[buf][0];
    unsigned wl = (unsigned)(uintptr_t)&Ws[buf][0];
    #pragma unroll
    for (int j = 0; j < 2; ++j) {                 // 128 rows x 64B
      int unit = tid + j * 256, row = unit >> 2, seg = unit & 3;
      async_b128(xl + row * 64 + seg * 16, xA,
                 (unsigned)(((m0 + row) * 1024 + k0) * 2 + seg * 16));
    }
    { int row = tid >> 2, seg = tid & 3;          // 64 rows x 64B
      async_b128(wl + row * 64 + seg * 16, wA,
                 (unsigned)(((n0 + row) * 1024 + k0) * 2 + seg * 16)); }
  };

  const v8f vzero = {0.f,0.f,0.f,0.f,0.f,0.f,0.f,0.f};
  v8f acc[2][2];
  acc[0][0] = vzero; acc[0][1] = vzero; acc[1][0] = vzero; acc[1][1] = vzero;

  issue(0, 0);
  for (int kt = 0; kt < 32; ++kt) {
    const int buf = kt & 1;
    wait_async0();                 // own slices of tile kt arrived
    __syncthreads();               // everyone's slices arrived; old buf free
    if (kt < 31) issue((kt + 1) * 32, buf ^ 1);

    v16bf a0 = ld_frag(&Xs[buf][(wm * 32 + 0  + ln) * 32], g, 0);
    v16bf a1 = ld_frag(&Xs[buf][(wm * 32 + 16 + ln) * 32], g, 0);
    v16bf b0 = ld_frag(&Ws[buf][(wn * 32 + 0  + ln) * 32], g, 0);
    v16bf b1 = ld_frag(&Ws[buf][(wn * 32 + 16 + ln) * 32], g, 0);
    acc[0][0] = WMMA_BF16(a0, b0, acc[0][0]);
    acc[0][1] = WMMA_BF16(a0, b1, acc[0][1]);
    acc[1][0] = WMMA_BF16(a1, b0, acc[1][0]);
    acc[1][1] = WMMA_BF16(a1, b1, acc[1][1]);
  }

  #pragma unroll
  for (int mi = 0; mi < 2; ++mi)
    #pragma unroll
    for (int nj = 0; nj < 2; ++nj)
      #pragma unroll
      for (int r = 0; r < 8; ++r)
        Cs[(wm * 32 + mi * 16 + g * 8 + r) * 64 + (wn * 32 + nj * 16 + ln)] =
            acc[mi][nj][r];
  __syncthreads();

  if (z < 2) {   // Q or K: bias + RoPE, store [B,H,L,Dh]
    __bf16* dst = (z == 0) ? Qb : Kb;
    const int c = tid & 63;
    for (int rr = tid >> 6; rr < 128; rr += 4) {
      int grow = m0 + rr, b = grow >> 11, pos = grow & 2047;
      float v  = Cs[rr * 64 + c]        + bi[n0 + c];
      float vp = Cs[rr * 64 + (c ^ 32)] + bi[n0 + (c ^ 32)];
      int j = c & 31;
      float invf = __powf(10000.f, -(float)j * (1.f / 32.f));
      float sth, cth;
      __sincosf((float)pos * invf, &sth, &cth);
      float rot = (c < 32) ? -vp : vp;
      float outv = v * cth + rot * sth;
      if (z == 0) outv *= 0.125f;
      dst[((size_t)((b * 16 + h) * 2048 + pos)) * 64 + c] = (__bf16)outv;
    }
  } else {       // V: bias only, store transposed [B,H,Dh,L] (pos-major coalesced)
    const int pr = tid & 31;
    for (int dd = tid >> 5; dd < 64; dd += 8)
      for (int rr = pr; rr < 128; rr += 32) {
        int grow = m0 + rr, b = grow >> 11, pos = grow & 2047;
        float outv = Cs[rr * 64 + dd] + bi[n0 + dd];
        Vb[((size_t)((b * 16 + h) * 64 + dd)) * 2048 + pos] = (__bf16)outv;
      }
  }
}

// ---------------------------------------------------------------------------
// Kernel 2: flash attention.  K [B,H,L,Dh] tiles and V [B,H,Dh,L] tiles are
// async-copied into double-buffered LDS; 1 barrier per key block.
// ---------------------------------------------------------------------------
__global__ __launch_bounds__(256) void flash_attn(
    const __bf16* __restrict__ Qb, const __bf16* __restrict__ Kb,
    const __bf16* __restrict__ Vb, __bf16* __restrict__ attn)
{
  __shared__ __bf16 Ks[2][64 * 64];   // [key][d]
  __shared__ __bf16 Vt[2][64 * 64];   // [d][key]
  __shared__ __bf16 Ps[128 * 64];     // P restage

  const int tid = threadIdx.x;
  const int wv = tid >> 5, lane = tid & 31, g = lane >> 4, ln = lane & 15;
  const int q0 = blockIdx.x * 128;
  const int bh = blockIdx.y;

  const unsigned long long kA = (unsigned long long)(uintptr_t)Kb;
  const unsigned long long vA = (unsigned long long)(uintptr_t)Vb;

  auto issue_kv = [&](int kb, int buf) {
    unsigned kl = (unsigned)(uintptr_t)&Ks[buf][0];
    unsigned vl = (unsigned)(uintptr_t)&Vt[buf][0];
    unsigned kgoff = (unsigned)((bh * 2048 + kb * 64) * 64 * 2); // contiguous 8KB
    #pragma unroll
    for (int j = 0; j < 2; ++j) {
      int unit = tid + j * 256;                       // 0..511
      async_b128(kl + unit * 16, kA, kgoff + unit * 16);
      int d = unit >> 3, seg = unit & 7;              // V: 64 rows x 128B
      async_b128(vl + d * 128 + seg * 16, vA,
                 (unsigned)(((bh * 64 + d) * 2048 + kb * 64) * 2 + seg * 16));
    }
  };

  const __bf16* Qrow = Qb + (size_t)(bh * 2048 + q0 + wv * 16 + ln) * 64;
  v16bf aQ0 = ld_frag(Qrow, g, 0);
  v16bf aQ1 = ld_frag(Qrow, g, 32);

  const v8f vzero = {0.f,0.f,0.f,0.f,0.f,0.f,0.f,0.f};
  v8f Oacc[4] = {vzero, vzero, vzero, vzero};
  float rowM[8], rowS[8];
  #pragma unroll
  for (int r = 0; r < 8; ++r) { rowM[r] = -1e30f; rowS[r] = 0.f; }

  issue_kv(0, 0);
  for (int kb = 0; kb < 32; ++kb) {
    const int buf = kb & 1;
    wait_async0();
    __syncthreads();
    if (kb < 31) issue_kv(kb + 1, buf ^ 1);

    // S = Q @ K^T
    v8f accS[4] = {vzero, vzero, vzero, vzero};
    #pragma unroll
    for (int cchunk = 0; cchunk < 2; ++cchunk) {
      v16bf a = cchunk ? aQ1 : aQ0;
      #pragma unroll
      for (int nj = 0; nj < 4; ++nj) {
        v16bf bK = ld_frag(&Ks[buf][(nj * 16 + ln) * 64], g, cchunk * 32);
        accS[nj] = WMMA_BF16(a, bK, accS[nj]);
      }
    }

    // Online softmax; lane owns rows m = r + 8g of its strip
    #pragma unroll
    for (int r = 0; r < 8; ++r) {
      float bm = fmaxf(fmaxf(accS[0][r], accS[1][r]),
                       fmaxf(accS[2][r], accS[3][r]));
      bm = fmaxf(bm, __shfl_xor(bm, 1, 32));
      bm = fmaxf(bm, __shfl_xor(bm, 2, 32));
      bm = fmaxf(bm, __shfl_xor(bm, 4, 32));
      bm = fmaxf(bm, __shfl_xor(bm, 8, 32));
      float mn = fmaxf(rowM[r], bm);
      float sc = __expf(rowM[r] - mn);
      rowM[r] = mn;
      float ps = 0.f;
      #pragma unroll
      for (int nj = 0; nj < 4; ++nj) {
        float p = __expf(accS[nj][r] - mn);
        accS[nj][r] = p;
        ps += p;
      }
      ps += __shfl_xor(ps, 1, 32);
      ps += __shfl_xor(ps, 2, 32);
      ps += __shfl_xor(ps, 4, 32);
      ps += __shfl_xor(ps, 8, 32);
      rowS[r] = rowS[r] * sc + ps;
      #pragma unroll
      for (int nj = 0; nj < 4; ++nj)
        Oacc[nj][r] = Oacc[nj][r] * sc;
    }

    // Restage P through the wave's private LDS strip (C-layout -> A-layout)
    #pragma unroll
    for (int nj = 0; nj < 4; ++nj)
      #pragma unroll
      for (int r = 0; r < 8; ++r)
        Ps[(wv * 16 + g * 8 + r) * 64 + nj * 16 + ln] = (__bf16)accS[nj][r];
    asm volatile("s_wait_dscnt 0" ::: "memory");

    // O += P @ V
    #pragma unroll
    for (int cchunk = 0; cchunk < 2; ++cchunk) {
      v16bf aP = ld_frag(&Ps[(wv * 16 + ln) * 64], g, cchunk * 32);
      #pragma unroll
      for (int nj = 0; nj < 4; ++nj) {
        v16bf bV = ld_frag(&Vt[buf][(nj * 16 + ln) * 64], g, cchunk * 32);
        Oacc[nj] = WMMA_BF16(aP, bV, Oacc[nj]);
      }
    }
  }

  const int b = bh >> 4, h = bh & 15;
  #pragma unroll
  for (int r = 0; r < 8; ++r) {
    float inv = 1.f / rowS[r];
    int pos = q0 + wv * 16 + g * 8 + r;
    size_t rowoff = (size_t)(b * 2048 + pos) * 1024;
    #pragma unroll
    for (int nj = 0; nj < 4; ++nj)
      attn[rowoff + h * 64 + nj * 16 + ln] = (__bf16)(Oacc[nj][r] * inv);
  }
}

// ---------------------------------------------------------------------------
// Kernel 3: out = attn @ Wo + bo  (bf16 A, async double-buffered, fp32 out)
// ---------------------------------------------------------------------------
__global__ __launch_bounds__(256) void out_proj_gemm(
    const __bf16* __restrict__ A, const __bf16* __restrict__ Wot,
    const float* __restrict__ bo, float* __restrict__ out)
{
  __shared__ __bf16 Xs[2][128 * 32];
  __shared__ __bf16 Ws[2][64 * 32];
  __shared__ float  Cs[128 * 64];

  const int tid = threadIdx.x;
  const int wv = tid >> 5, lane = tid & 31, g = lane >> 4, ln = lane & 15;
  const int wm = wv & 3, wn = wv >> 2;
  const int m0 = blockIdx.x * 128;
  const int n0 = blockIdx.y * 64;

  const unsigned long long aA = (unsigned long long)(uintptr_t)A;
  const unsigned long long wA = (unsigned long long)(uintptr_t)Wot;

  auto issue = [&](int k0, int buf) {
    unsigned xl = (unsigned)(uintptr_t)&Xs[buf][0];
    unsigned wl = (unsigned)(uintptr_t)&Ws[buf][0];
    #pragma unroll
    for (int j = 0; j < 2; ++j) {
      int unit = tid + j * 256, row = unit >> 2, seg = unit & 3;
      async_b128(xl + row * 64 + seg * 16, aA,
                 (unsigned)(((m0 + row) * 1024 + k0) * 2 + seg * 16));
    }
    { int row = tid >> 2, seg = tid & 3;
      async_b128(wl + row * 64 + seg * 16, wA,
                 (unsigned)(((n0 + row) * 1024 + k0) * 2 + seg * 16)); }
  };

  const v8f vzero = {0.f,0.f,0.f,0.f,0.f,0.f,0.f,0.f};
  v8f acc[2][2];
  acc[0][0] = vzero; acc[0][1] = vzero; acc[1][0] = vzero; acc[1][1] = vzero;

  issue(0, 0);
  for (int kt = 0; kt < 32; ++kt) {
    const int buf = kt & 1;
    wait_async0();
    __syncthreads();
    if (kt < 31) issue((kt + 1) * 32, buf ^ 1);

    v16bf a0 = ld_frag(&Xs[buf][(wm * 32 + 0  + ln) * 32], g, 0);
    v16bf a1 = ld_frag(&Xs[buf][(wm * 32 + 16 + ln) * 32], g, 0);
    v16bf b0 = ld_frag(&Ws[buf][(wn * 32 + 0  + ln) * 32], g, 0);
    v16bf b1 = ld_frag(&Ws[buf][(wn * 32 + 16 + ln) * 32], g, 0);
    acc[0][0] = WMMA_BF16(a0, b0, acc[0][0]);
    acc[0][1] = WMMA_BF16(a0, b1, acc[0][1]);
    acc[1][0] = WMMA_BF16(a1, b0, acc[1][0]);
    acc[1][1] = WMMA_BF16(a1, b1, acc[1][1]);
  }

  #pragma unroll
  for (int mi = 0; mi < 2; ++mi)
    #pragma unroll
    for (int nj = 0; nj < 2; ++nj)
      #pragma unroll
      for (int r = 0; r < 8; ++r)
        Cs[(wm * 32 + mi * 16 + g * 8 + r) * 64 + (wn * 32 + nj * 16 + ln)] =
            acc[mi][nj][r];
  __syncthreads();

  const int c = tid & 63;
  for (int rr = tid >> 6; rr < 128; rr += 4)
    out[(size_t)(m0 + rr) * 1024 + n0 + c] = Cs[rr * 64 + c] + bo[n0 + c];
}

// ---------------------------------------------------------------------------
extern "C" void kernel_launch(void* const* d_in, const int* in_sizes, int n_in,
                              void* d_out, int out_size, void* d_ws, size_t ws_size,
                              hipStream_t stream) {
  const float* x  = (const float*)d_in[0];
  const float* Wq = (const float*)d_in[1];
  const float* bq = (const float*)d_in[2];
  const float* Wk = (const float*)d_in[3];
  const float* bk = (const float*)d_in[4];
  const float* Wv = (const float*)d_in[5];
  const float* bv = (const float*)d_in[6];
  const float* Wo = (const float*)d_in[7];
  const float* bo = (const float*)d_in[8];
  float* out = (float*)d_out;

  char* ws = (char*)d_ws;
  const size_t MB = 1024 * 1024;
  __bf16* xbf  = (__bf16*)(ws + 0 * MB);    // 8 MB
  __bf16* Wqt  = (__bf16*)(ws + 8 * MB);    // 2 MB each
  __bf16* Wkt  = (__bf16*)(ws + 10 * MB);
  __bf16* Wvt  = (__bf16*)(ws + 12 * MB);
  __bf16* Wot  = (__bf16*)(ws + 14 * MB);
  __bf16* Qb   = (__bf16*)(ws + 16 * MB);   // 8 MB each
  __bf16* Kb   = (__bf16*)(ws + 24 * MB);
  __bf16* Vb   = (__bf16*)(ws + 32 * MB);   // transposed [B,H,Dh,L]
  __bf16* attn = (__bf16*)(ws + 40 * MB);

  convert_x<<<4096, 256, 0, stream>>>(x, xbf);
  transpose_w<<<dim3(16, 16, 4), 256, 0, stream>>>(Wq, Wk, Wv, Wo,
                                                   Wqt, Wkt, Wvt, Wot);
  qkv_rope_gemm<<<dim3(32, 16, 3), 256, 0, stream>>>(
      xbf, Wqt, bq, Wkt, bk, Wvt, bv, Qb, Kb, Vb);
  flash_attn<<<dim3(16, 32), 256, 0, stream>>>(Qb, Kb, Vb, attn);
  out_proj_gemm<<<dim3(32, 16), 256, 0, stream>>>(attn, Wot, bo, out);
}
